// ParityViolationEGNN_43782896615991
// MI455X (gfx1250) — matrix operand order repaired and checked by reading, not compiled, not run on hardware
//
#include <hip/hip_runtime.h>

typedef __attribute__((ext_vector_type(16))) _Float16 v16h;
typedef __attribute__((ext_vector_type(8)))  _Float16 v8h;
typedef __attribute__((ext_vector_type(2)))  __fp16   v2hf;   // return type of cvt_pkrtz
typedef __attribute__((ext_vector_type(8)))  float    v8f;
typedef __attribute__((ext_vector_type(4)))  unsigned int v4u;

#define HDIM 16
#define LDS_COLS 120           // padded row stride in f16 (240B; 60-bank step, conflict-free)
#define ONES_CHUNK 10          // cols 80..87   = [1,0,...,0]  (bias-folding)
#define ZERO_CHUNK 11          // cols 88..95   = zeros        (unconditional pad reads)
#define IN0_CHUNK  12          // cols 96..103  = node0 raw input
#define IN1_CHUNK  13          // cols 104..111 = node1 raw input
#define INE_CHUNK  14          // cols 112..119 = edge raw input
#define WAVES_PER_BLOCK 8
#define TILES_PER_WAVE 16      // amortize the weight-fragment prologue further
#define ELEMS_PER_BLOCK (16 * WAVES_PER_BLOCK * TILES_PER_WAVE)   // 2048

// ---- D = A(16x32 f16) * B(32x16 f16) + C(16x16 f32) ----
static __device__ __forceinline__ v8f wmma16(v16h a, v16h b, v8f c) {
  return __builtin_amdgcn_wmma_f32_16x16x32_f16(
      false, a, false, b, (short)0, c, false, false);
}

// Single-instruction ReLU (avoids fmaxf's canonicalize+max pair).
static __device__ __forceinline__ float relu1(float x) {
  float r;
  asm("v_max_num_f32 %0, 0, %1" : "=v"(r) : "v"(x));
  return r;
}

// B fragment for W (in_dim x 16 fp32, row-major), no bias.
static __device__ __forceinline__ v16h bfrag(const float* __restrict__ W,
                                             int in_dim, int col, int kbase) {
  v16h b;
#pragma unroll
  for (int e = 0; e < 16; ++e) {
    int k = (kbase >> 1) + e + (e >= 8 ? 8 : 0);
    b[e] = (k < in_dim) ? (_Float16)W[k * HDIM + col] : (_Float16)0.0f;
  }
  return b;
}

// B fragment with bias folded in at row K == in_dim (pairs with the LDS ones-chunk
// or a staged 1.0 in the input chunk); rows optionally scaled.
static __device__ __forceinline__ v16h bfragb(const float* __restrict__ W,
                                              int in_dim,
                                              const float* __restrict__ bias,
                                              int col, int kbase, float scale) {
  v16h b;
#pragma unroll
  for (int e = 0; e < 16; ++e) {
    int k = (kbase >> 1) + e + (e >= 8 ? 8 : 0);
    float v = (k < in_dim) ? W[k * HDIM + col] * scale
                           : ((k == in_dim) ? bias[col] : 0.0f);
    b[e] = (_Float16)v;
  }
  return b;
}

// A fragment (16x32 f16 layout): two unconditional 16B chunk loads from this
// lane's LDS row. (pl0,pl1)/(ph0,ph1) = physical chunk for half 0/1 of the
// lo/hi K-group; pass ZERO_CHUNK for padding.
static __device__ __forceinline__ v16h ldsA(const _Float16* rowp,
                                            int pl0, int pl1, int ph0, int ph1,
                                            int hsel) {
  int pl = hsel ? pl1 : pl0;
  int ph = hsel ? ph1 : ph0;
  v8h lo = *(const v8h*)(rowp + pl * 8);   // ds_load_b128
  v8h hi = *(const v8h*)(rowp + ph * 8);   // ds_load_b128
  v16h a;
#pragma unroll
  for (int i = 0; i < 8; ++i) { a[i] = lo[i]; a[8 + i] = hi[i]; }
  return a;
}

// Stage a C fragment into LDS as f16 (optional ReLU), packed f32->f16 cvt.
static __device__ __forceinline__ void stage16(_Float16* wlds, int colOff, v8f c,
                                               bool relu, int mbase, int col) {
#pragma unroll
  for (int i = 0; i < 4; ++i) {
    float a = c[2 * i], b = c[2 * i + 1];
    if (relu) { a = relu1(a); b = relu1(b); }
    v2hf p = __builtin_amdgcn_cvt_pkrtz(a, b);              // v_cvt_pk_f16_f32
    wlds[(mbase + 2 * i)     * LDS_COLS + colOff + col] = (_Float16)p[0];
    wlds[(mbase + 2 * i + 1) * LDS_COLS + colOff + col] = (_Float16)p[1];
  }
}

// Same, with explicit bias add + ReLU (only the 32-dim upd_W1 stage needs it).
static __device__ __forceinline__ void stage16br(_Float16* wlds, int colOff, v8f c,
                                                 float bias, int mbase, int col) {
#pragma unroll
  for (int i = 0; i < 4; ++i) {
    float a = relu1(c[2 * i] + bias);
    float b = relu1(c[2 * i + 1] + bias);
    v2hf p = __builtin_amdgcn_cvt_pkrtz(a, b);
    wlds[(mbase + 2 * i)     * LDS_COLS + colOff + col] = (_Float16)p[0];
    wlds[(mbase + 2 * i + 1) * LDS_COLS + colOff + col] = (_Float16)p[1];
  }
}

__global__ __launch_bounds__(WAVES_PER_BLOCK * 32)
void egnn_wmma_kernel(
    const float* __restrict__ node_features,
    const float* __restrict__ e_dist,
    const float* __restrict__ e_dz,
    const float* __restrict__ e_sphi,
    const float* __restrict__ neW1, const float* __restrict__ neb1,
    const float* __restrict__ neW2, const float* __restrict__ neb2,
    const float* __restrict__ eeW1, const float* __restrict__ eeb1,
    const float* __restrict__ eeW2, const float* __restrict__ eeb2,
    const float* __restrict__ msgW1, const float* __restrict__ msgb1,
    const float* __restrict__ msgW2, const float* __restrict__ msgb2,
    const float* __restrict__ updW1, const float* __restrict__ updb1,
    const float* __restrict__ updW2, const float* __restrict__ updb2,
    const float* __restrict__ clsW1, const float* __restrict__ clsb1,
    const float* __restrict__ clsW2, const float* __restrict__ clsb2,
    float* __restrict__ out, int Btot) {

  __shared__ __align__(16) _Float16 lds[WAVES_PER_BLOCK * 16 * LDS_COLS];

  const int lane  = threadIdx.x & 31;
  const int wave  = threadIdx.x >> 5;
  const int row   = lane & 15;
  const int hsel  = lane >> 4;
  const int col   = row;
  const int kbase = hsel * 16;
  const int mbase = hsel * 8;

  _Float16* wlds  = lds + wave * 16 * LDS_COLS;
  _Float16* myrow = wlds + row * LDS_COLS;

  // Constant chunks: ONES = [1,0,...,0], ZERO = all zeros.
  {
    v8h ones = {};
    ones[0] = (_Float16)1.0f;
    *(v8h*)(myrow + ONES_CHUNK * 8) = ones;
    v8h zer = {};
    *(v8h*)(myrow + ZERO_CHUNK * 8) = zer;
  }

  // ---- Weight B-fragments (biases folded at K == in_dim where possible) ----
  v16h B_ne1 = bfragb(neW1, 2,  neb1, col, kbase, 1.0f);
  v16h B_ne2 = bfragb(neW2, 16, neb2, col, kbase, 1.0f);
  v16h B_ee1 = bfragb(eeW1, 3,  eeb1, col, kbase, 1.0f);
  v16h B_ee2 = bfragb(eeW2, 16, eeb2, col, kbase, 1.0f);
  v16h B_m1a[2], B_m1b[2], B_m2[2], B_u1[2], B_u2[2];
  float bu1[2];
#pragma unroll
  for (int l = 0; l < 2; ++l) {
    const float* w1 = msgW1 + l * 48 * HDIM;
    B_m1a[l] = bfrag (w1, 48, col, kbase);                                  // K 0..31
    B_m1b[l] = bfragb(w1 + 32 * HDIM, 16, msgb1 + l * HDIM, col, kbase, 1.0f); // +bias@48
    B_m2[l]  = bfragb(msgW2 + l * 256, 16, msgb2 + l * HDIM, col, kbase, 1.0f);
    B_u1[l]  = bfrag (updW1 + l * 32 * HDIM, 32, col, kbase);               // bias explicit
    B_u2[l]  = bfragb(updW2 + l * 256, 16, updb2 + l * HDIM, col, kbase, 1.0f);
    bu1[l]   = updb1[l * HDIM + col];
  }
  // mean(h0,h1) folded: scale cls_W1 rows by 0.5 (bias row unscaled).
  v16h B_c1 = bfragb(clsW1, 16, clsb1, col, kbase, 0.5f);
  // Final 16->1 dot as a WMMA: cls_W2 (+cls_b2 at K=16) packed into column 0.
  v16h B_c2;
#pragma unroll
  for (int e = 0; e < 16; ++e) {
    int k = (kbase >> 1) + e + (e >= 8 ? 8 : 0);
    float v = 0.0f;
    if (col == 0) v = (k < 16) ? clsW2[k] : ((k == 16) ? clsb2[0] : 0.0f);
    B_c2[e] = (_Float16)v;
  }

  const long long waveBase =
      ((long long)blockIdx.x * WAVES_PER_BLOCK + wave) * (TILES_PER_WAVE * 16LL);

  for (int t = 0; t < TILES_PER_WAVE; ++t) {
    const long long base = waveBase + (long long)t * 16;
    long long gi = base + row;
    if (gi >= Btot) gi = Btot - 1;                 // clamp: keep EXEC full for WMMA
    if (t + 1 < TILES_PER_WAVE)
      __builtin_prefetch(node_features + (base + 16 + row) * 4, 0, 3);

    const float* nf = node_features + gi * 4;
    const float f0 = nf[0], f1 = nf[1], f2 = nf[2], f3 = nf[3];
    const float ed = e_dist[gi], ez = e_dz[gi], ep = e_sphi[gi];

    // ---- Stage all three raw inputs (independent b128 stores) ------------
    {
      v2hf p = __builtin_amdgcn_cvt_pkrtz(f0, f1);
      v4u st = {__builtin_bit_cast(unsigned int, p), 0x00003C00u, 0u, 0u};
      *(v4u*)(myrow + IN0_CHUNK * 8) = st;                  // [f0,f1,1,0...]
    }
    {
      v2hf p = __builtin_amdgcn_cvt_pkrtz(f2, f3);
      v4u st = {__builtin_bit_cast(unsigned int, p), 0x00003C00u, 0u, 0u};
      *(v4u*)(myrow + IN1_CHUNK * 8) = st;                  // [f2,f3,1,0...]
    }
    {
      v2hf p0 = __builtin_amdgcn_cvt_pkrtz(ed, ez);
      v2hf p1 = __builtin_amdgcn_cvt_pkrtz(ep, 1.0f);
      v4u st = {__builtin_bit_cast(unsigned int, p0),
                __builtin_bit_cast(unsigned int, p1), 0u, 0u};
      *(v4u*)(myrow + INE_CHUNK * 8) = st;                  // [ed,ez,ep,1,0...]
    }

    // ---- Embeds: three independent 2-matmul chains, interleaved ----------
    v16h aN0 = ldsA(myrow, IN0_CHUNK, ZERO_CHUNK, ZERO_CHUNK, ZERO_CHUNK, hsel);
    v16h aN1 = ldsA(myrow, IN1_CHUNK, ZERO_CHUNK, ZERO_CHUNK, ZERO_CHUNK, hsel);
    v16h aEE = ldsA(myrow, INE_CHUNK, ZERO_CHUNK, ZERO_CHUNK, ZERO_CHUNK, hsel);
    v8f c0 = {}; c0 = wmma16(aN0, B_ne1, c0);
    v8f c1 = {}; c1 = wmma16(aN1, B_ne1, c1);
    v8f c2 = {}; c2 = wmma16(aEE, B_ee1, c2);
    stage16(wlds, 48, c0, true, mbase, col);                // n0 hidden -> 6,7
    stage16(wlds, 64, c1, true, mbase, col);                // n1 hidden -> 8,9
    stage16(wlds, 0,  c2, true, mbase, col);                // e  hidden -> 0,1 (temp)
    v16h hA = ldsA(myrow, 6, 7, ONES_CHUNK, ZERO_CHUNK, hsel);
    v16h hB = ldsA(myrow, 8, 9, ONES_CHUNK, ZERO_CHUNK, hsel);
    v16h hC = ldsA(myrow, 0, 1, ONES_CHUNK, ZERO_CHUNK, hsel);
    v8f h0C = {}; h0C = wmma16(hA, B_ne2, h0C);             // bias folded
    v8f h1C = {}; h1C = wmma16(hB, B_ne2, h1C);
    c2 = (v8f){};  c2  = wmma16(hC, B_ee2, c2);
    stage16(wlds, 0,  h0C, false, mbase, col);              // h0 -> 0,1 (after hC read)
    stage16(wlds, 16, h1C, false, mbase, col);              // h1 -> 2,3
    stage16(wlds, 32, c2,  false, mbase, col);              // e  -> 4,5

    // ---- Message-passing layers (paired independent chains) --------------
#pragma unroll
    for (int l = 0; l < 2; ++l) {
      // Both msg hiddens first (the [e|1] fragment is shared).
      v16h aM01 = ldsA(myrow, 0, 1, 2, 3, hsel);
      v16h aM10 = ldsA(myrow, 2, 3, 0, 1, hsel);
      v16h aEb  = ldsA(myrow, 4, 5, ONES_CHUNK, ZERO_CHUNK, hsel);
      v8f cH01 = {}; cH01 = wmma16(aM01, B_m1a[l], cH01);
      v8f cH10 = {}; cH10 = wmma16(aM10, B_m1a[l], cH10);
      cH01 = wmma16(aEb, B_m1b[l], cH01);
      cH10 = wmma16(aEb, B_m1b[l], cH10);
      stage16(wlds, 48, cH01, true, mbase, col);            // -> 6,7
      stage16(wlds, 64, cH10, true, mbase, col);            // -> 8,9

      // Both msg outputs.
      v16h m01 = ldsA(myrow, 6, 7, ONES_CHUNK, ZERO_CHUNK, hsel);
      v16h m10 = ldsA(myrow, 8, 9, ONES_CHUNK, ZERO_CHUNK, hsel);
      v8f cM01 = {}; cM01 = wmma16(m01, B_m2[l], cM01);
      v8f cM10 = {}; cM10 = wmma16(m10, B_m2[l], cM10);
      stage16(wlds, 48, cM01, false, mbase, col);           // msg01 -> 6,7
      stage16(wlds, 64, cM10, false, mbase, col);           // msg10 -> 8,9

      // Both upd hiddens.
      v16h aU0 = ldsA(myrow, 0, 1, 6, 7, hsel);             // [h0|msg01]
      v16h aU1 = ldsA(myrow, 2, 3, 8, 9, hsel);             // [h1|msg10]
      v8f cU0 = {}; cU0 = wmma16(aU0, B_u1[l], cU0);
      v8f cU1 = {}; cU1 = wmma16(aU1, B_u1[l], cU1);
      stage16br(wlds, 48, cU0, bu1[l], mbase, col);         // -> 6,7
      stage16br(wlds, 64, cU1, bu1[l], mbase, col);         // -> 8,9

      // Both h accumulations (C-init = h; bias folded in B_u2).
      v16h uH0 = ldsA(myrow, 6, 7, ONES_CHUNK, ZERO_CHUNK, hsel);
      v16h uH1 = ldsA(myrow, 8, 9, ONES_CHUNK, ZERO_CHUNK, hsel);
      h0C = wmma16(uH0, B_u2[l], h0C);
      h1C = wmma16(uH1, B_u2[l], h1C);
      if (l == 0) {                                         // last layer: LDS h is dead
        stage16(wlds, 0,  h0C, false, mbase, col);
        stage16(wlds, 16, h1C, false, mbase, col);
      }
    }

    // ---- Classifier (mean's 0.5 folded into B_c1) ------------------------
    v8f g8;
#pragma unroll
    for (int r = 0; r < 8; ++r) g8[r] = h0C[r] + h1C[r];
    stage16(wlds, 48, g8, false, mbase, col);               // (h0+h1) -> 6,7
    v16h gA = ldsA(myrow, 6, 7, ONES_CHUNK, ZERO_CHUNK, hsel);
    v8f cc = {}; cc = wmma16(gA, B_c1, cc);
    stage16(wlds, 64, cc, true, mbase, col);                // cls hidden -> 8,9
    v16h gB = ldsA(myrow, 8, 9, ONES_CHUNK, ZERO_CHUNK, hsel);
    cc = (v8f){}; cc = wmma16(gB, B_c2, cc);                // 28th WMMA: 16->1 dot (+bias)

    // Column 0 of C holds the logits; stage as f32 and fan out one per row.
#pragma unroll
    for (int r = 0; r < 8; ++r)
      *(float*)(wlds + (mbase + r) * LDS_COLS + 48 + 2 * col) = cc[r];
    float logit = *(const float*)(myrow + 48);
    if (lane < 16 && (base + row) < Btot) out[base + row] = logit;
  }
}

extern "C" void kernel_launch(void* const* d_in, const int* in_sizes, int n_in,
                              void* d_out, int out_size, void* d_ws, size_t ws_size,
                              hipStream_t stream) {
  const int Btot = in_sizes[1];   // edge_distance_3d has B elements
  const float* nf   = (const float*)d_in[0];
  const float* ed   = (const float*)d_in[1];
  const float* ez   = (const float*)d_in[2];
  const float* ep   = (const float*)d_in[3];
  const float* neW1 = (const float*)d_in[4];  const float* neb1 = (const float*)d_in[5];
  const float* neW2 = (const float*)d_in[6];  const float* neb2 = (const float*)d_in[7];
  const float* eeW1 = (const float*)d_in[8];  const float* eeb1 = (const float*)d_in[9];
  const float* eeW2 = (const float*)d_in[10]; const float* eeb2 = (const float*)d_in[11];
  const float* msgW1= (const float*)d_in[12]; const float* msgb1= (const float*)d_in[13];
  const float* msgW2= (const float*)d_in[14]; const float* msgb2= (const float*)d_in[15];
  const float* updW1= (const float*)d_in[16]; const float* updb1= (const float*)d_in[17];
  const float* updW2= (const float*)d_in[18]; const float* updb2= (const float*)d_in[19];
  const float* clsW1= (const float*)d_in[20]; const float* clsb1= (const float*)d_in[21];
  const float* clsW2= (const float*)d_in[22]; const float* clsb2= (const float*)d_in[23];

  const int grid = (Btot + ELEMS_PER_BLOCK - 1) / ELEMS_PER_BLOCK;
  egnn_wmma_kernel<<<grid, WAVES_PER_BLOCK * 32, 0, stream>>>(
      nf, ed, ez, ep,
      neW1, neb1, neW2, neb2, eeW1, eeb1, eeW2, eeb2,
      msgW1, msgb1, msgW2, msgb2, updW1, updb1, updW2, updb2,
      clsW1, clsb1, clsW2, clsb2,
      (float*)d_out, Btot);
}
